// TFT_NoVSN_49108656063294
// MI455X (gfx1250) — matrix operand order repaired and checked by reading, not compiled, not run on hardware
//
#include <hip/hip_runtime.h>
#include <cstddef>

typedef __attribute__((ext_vector_type(16))) __bf16          v16bf;
typedef __attribute__((ext_vector_type(8)))  float           v8f;
typedef __attribute__((ext_vector_type(8)))  unsigned short  v8us;
typedef __attribute__((ext_vector_type(16))) unsigned short  v16us;
typedef __attribute__((ext_vector_type(8)))  unsigned int    v8ui;
typedef __attribute__((ext_vector_type(2)))  __bf16          v2bf;
typedef __attribute__((ext_vector_type(4)))  int             v4i;

#if defined(__HIP_DEVICE_COMPILE__) && __has_builtin(__builtin_amdgcn_global_load_async_to_lds_b128)
#define HAVE_ASYNC 1
typedef __attribute__((address_space(1))) v4i* as1_v4i;
typedef __attribute__((address_space(3))) v4i* as3_v4i;
#else
#define HAVE_ASYNC 0
#endif
#if defined(__HIP_DEVICE_COMPILE__) && __has_builtin(__builtin_amdgcn_cvt_pk_bf16_f32)
#define HAVE_PK 1
#else
#define HAVE_PK 0
#endif

// ---------------- helpers ----------------

__device__ __forceinline__ unsigned short f2bf(float f) {
  unsigned u = __float_as_uint(f);
  u += 0x7FFFu + ((u >> 16) & 1u);  // round-to-nearest-even
  return (unsigned short)(u >> 16);
}

// pack two floats -> two bf16 in one dword
__device__ __forceinline__ unsigned cvt2(float lo, float hi) {
#if HAVE_PK
  v2bf r = __builtin_amdgcn_cvt_pk_bf16_f32(lo, hi);
  return __builtin_bit_cast(unsigned, r);
#else
  return (unsigned)f2bf(lo) | ((unsigned)f2bf(hi) << 16);
#endif
}

__device__ __forceinline__ v16bf frag2(const unsigned short* p0, const unsigned short* p1) {
  v8us a = *(const v8us*)p0;
  v8us b = *(const v8us*)p1;
  v16us t;
#pragma unroll
  for (int i = 0; i < 8; i++) { t[i] = a[i]; t[i + 8] = b[i]; }
  return __builtin_bit_cast(v16bf, t);
}

__device__ __forceinline__ v8f wmma_bf16(v16bf a, v16bf b, v8f c) {
  return __builtin_amdgcn_wmma_f32_16x16x32_bf16(false, a, false, b, (short)0, c, false, false);
}

__device__ __forceinline__ float wred(float v) {
#pragma unroll
  for (int off = 16; off > 0; off >>= 1) v += __shfl_xor(v, off, 32);
  return v;
}

// copy 16B from global (bf16 weights) into LDS, async if available
__device__ __forceinline__ void lds_fill16(const unsigned short* gp, unsigned short* lp) {
#if HAVE_ASYNC
  __builtin_amdgcn_global_load_async_to_lds_b128(
      (as1_v4i)(v4i*)(unsigned short*)gp,
      (as3_v4i)(v4i*)lp, 0, 0);
#else
  *(v8us*)lp = *(const v8us*)gp;
#endif
}

__device__ __forceinline__ void lds_fill_fence() {
#if HAVE_ASYNC
  asm volatile("s_wait_asynccnt 0" ::: "memory");
#endif
}

// ---------------- weight convert+transpose: Wt[n][kp] = bf16(W[kp][n]) ----------------

__global__ __launch_bounds__(256) void wconv_kernel(const float* __restrict__ W,
                                                    unsigned short* __restrict__ Wt,
                                                    int K, int N, int Kp) {
  int e = blockIdx.x * 256 + threadIdx.x;
  if (e >= N * Kp) return;
  int n = e / Kp, kp = e - n * Kp;
  float v = (kp < K) ? W[(size_t)kp * N + n] : 0.f;
  Wt[(size_t)n * Kp + kp] = f2bf(v);
}

// ---------------- GEMM: C = act(A[MxK] @ W[KxN] + bias) (+PE) ----------------
// W pre-transposed bf16 (N x Kp). block: 256 thr = 8 waves, tile 64x128, K-step 32.

template <int ACT, int ADDPE>
__global__ __launch_bounds__(256) void gemm_kernel(
    const float* __restrict__ A, const unsigned short* __restrict__ Wt,
    const float* __restrict__ bias, float* __restrict__ C,
    int N, int K, int Kp, const float* __restrict__ pe, int Sper) {
  __shared__ __align__(16) unsigned short a_lds[64][32];    // 4 KB
  __shared__ __align__(16) unsigned short bt_lds[128][32];  // 8 KB

  const int tid   = threadIdx.x;
  const int wave  = tid >> 5;
  const int lane  = tid & 31;
  const int lmod  = lane & 15;
  const int lhalf = lane >> 4;
  const int kbase = lhalf * 8;
  const int m0    = blockIdx.x * 64;
  const int n0    = blockIdx.y * 128;
  const int mSub  = wave & 3;
  const int nGrp  = wave >> 2;

  v8f acc[4] = {};

  const int amm = tid >> 2;       // 0..63 : A row
  const int akk = (tid & 3) * 8;  // 0/8/16/24 : A col chunk

  for (int k0 = 0; k0 < Kp; k0 += 32) {
    // A tile 64x32: each thread converts 8 contiguous row elements
    {
      const float* ap = A + (size_t)(m0 + amm) * K + k0 + akk;
      float f[8];
#pragma unroll
      for (int j = 0; j < 8; j++) f[j] = (k0 + akk + j < K) ? ap[j] : 0.f;
      uint4 st;
      st.x = cvt2(f[0], f[1]); st.y = cvt2(f[2], f[3]);
      st.z = cvt2(f[4], f[5]); st.w = cvt2(f[6], f[7]);
      *(uint4*)&a_lds[amm][akk] = st;
    }
    // B tile 128x32 bf16: raw 16B chunks, async -> LDS when available
#pragma unroll
    for (int j = 0; j < 2; j++) {
      int c = j * 256 + tid;  // 512 chunks of 16B
      int nn = c >> 2, cc = c & 3;
      lds_fill16(Wt + (size_t)(n0 + nn) * Kp + k0 + cc * 8, &bt_lds[nn][cc * 8]);
    }
    lds_fill_fence();
    __syncthreads();

    const unsigned short* ap = &a_lds[mSub * 16 + lmod][0];
    v16bf afrag = frag2(ap + kbase, ap + 16 + kbase);
#pragma unroll
    for (int t = 0; t < 4; t++) {
      const unsigned short* bp = &bt_lds[nGrp * 64 + t * 16 + lmod][kbase];
      acc[t] = wmma_bf16(afrag, frag2(bp, bp + 16), acc[t]);
    }
    __syncthreads();
  }

#pragma unroll
  for (int t = 0; t < 4; t++) {
    int col = n0 + nGrp * 64 + t * 16 + lmod;
    float bb = bias[col];
#pragma unroll
    for (int r = 0; r < 8; r++) {
      int row = m0 + mSub * 16 + r + 8 * lhalf;
      float v = acc[t][r] + bb;
      if (ADDPE) v += pe[(size_t)(row % Sper) * N + col];
      if (ACT == 1) v = (v > 0.f) ? v : (__expf(v) - 1.f);  // ELU
      C[(size_t)row * N + col] = v;
    }
  }
}

// ---------------- attention: one wave per (b, head, 16 q-rows) ----------------

#define S_ 512
#define H_ 256
#define NH_ 8
#define DH_ 32

__global__ __launch_bounds__(32) void attn_kernel(
    const float* __restrict__ Qg, const float* __restrict__ Kg,
    const float* __restrict__ Vg, float* __restrict__ ctx,
    float* __restrict__ aw) {
  __shared__ __align__(16) float          sc[16][S_];   // 32 KB scores
  __shared__ __align__(16) unsigned short pb[16][S_];   // 16 KB probs bf16
  __shared__ __align__(16) unsigned short vt[DH_][32];  // 2 KB V^T tile

  const int lane  = threadIdx.x;
  const int qt    = blockIdx.x & 31;
  const int h     = (blockIdx.x >> 5) & 7;
  const int b     = blockIdx.x >> 8;
  const int lmod  = lane & 15;
  const int lhalf = lane >> 4;
  const int kbase = lhalf * 8;
  const float scale = 0.17677669529663687f;  // 1/sqrt(32)

  // Q fragment (A-matrix 16x32), scale folded in
  const float* qrow = Qg + ((size_t)(b * S_ + qt * 16 + lmod) * H_) + h * DH_;
  v8ui qw;
  {
    const float4* pa = (const float4*)(qrow + kbase);
    const float4* pc = (const float4*)(qrow + 16 + kbase);
    float4 a0 = pa[0], a1 = pa[1], c0 = pc[0], c1 = pc[1];
    qw[0] = cvt2(a0.x * scale, a0.y * scale); qw[1] = cvt2(a0.z * scale, a0.w * scale);
    qw[2] = cvt2(a1.x * scale, a1.y * scale); qw[3] = cvt2(a1.z * scale, a1.w * scale);
    qw[4] = cvt2(c0.x * scale, c0.y * scale); qw[5] = cvt2(c0.z * scale, c0.w * scale);
    qw[6] = cvt2(c1.x * scale, c1.y * scale); qw[7] = cvt2(c1.z * scale, c1.w * scale);
  }
  v16bf qfrag = __builtin_bit_cast(v16bf, qw);

  // scores = Q K^T, masked, into LDS
  for (int kt = 0; kt < S_ / 16; kt++) {
    const float* krow = Kg + ((size_t)(b * S_ + kt * 16 + lmod) * H_) + h * DH_ + kbase;
    const float4* k4 = (const float4*)krow;
    float4 k0v = k4[0], k1v = k4[1], k2v = k4[4], k3v = k4[5];
    v8ui kw;
    kw[0] = cvt2(k0v.x, k0v.y); kw[1] = cvt2(k0v.z, k0v.w);
    kw[2] = cvt2(k1v.x, k1v.y); kw[3] = cvt2(k1v.z, k1v.w);
    kw[4] = cvt2(k2v.x, k2v.y); kw[5] = cvt2(k2v.z, k2v.w);
    kw[6] = cvt2(k3v.x, k3v.y); kw[7] = cvt2(k3v.z, k3v.w);
    v8f s = {};
    s = wmma_bf16(qfrag, __builtin_bit_cast(v16bf, kw), s);
    int kcol = kt * 16 + lmod;
#pragma unroll
    for (int r = 0; r < 8; r++) {
      int qr = qt * 16 + r + 8 * lhalf;
      sc[r + 8 * lhalf][kcol] = (kcol > qr) ? -1e30f : s[r];
    }
  }
  __syncthreads();

  // softmax: lane handles row l%16, half (l/16)*256
  {
    int row = lmod, base = lhalf * 256;
    float mx = -1e30f;
    for (int c = 0; c < 256; c++) mx = fmaxf(mx, sc[row][base + c]);
    mx = fmaxf(mx, __shfl_xor(mx, 16, 32));
    float sum = 0.f;
    for (int c = 0; c < 256; c++) sum += __expf(sc[row][base + c] - mx);
    sum += __shfl_xor(sum, 16, 32);
    float inv = 1.f / sum;
    int qr = qt * 16 + row;
    float* awrow = aw ? aw + ((((size_t)b * NH_ + h) * S_ + qr) * S_) : nullptr;
    for (int c = 0; c < 256; c++) {
      float p = __expf(sc[row][base + c] - mx) * inv;
      pb[row][base + c] = f2bf(p);
      if (awrow) awrow[base + c] = p;
    }
  }
  __syncthreads();

  // ctx = P @ V
  v8f c0 = {}, c1 = {};
  for (int kt = 0; kt < S_ / 32; kt++) {
    const float* vrow = Vg + ((size_t)(b * S_ + kt * 32 + lane) * H_) + h * DH_;
#pragma unroll
    for (int d = 0; d < DH_; d++) vt[d][lane] = f2bf(vrow[d]);
    __syncthreads();

    int kb = kt * 32;
    const unsigned short* pp = &pb[lmod][kb];
    v16bf pfrag = frag2(pp + kbase, pp + 16 + kbase);
    const unsigned short* v0 = &vt[lmod][kbase];
    const unsigned short* v1 = &vt[16 + lmod][kbase];
    c0 = wmma_bf16(pfrag, frag2(v0, v0 + 16), c0);
    c1 = wmma_bf16(pfrag, frag2(v1, v1 + 16), c1);
    __syncthreads();
  }

#pragma unroll
  for (int r = 0; r < 8; r++) {
    int qr = qt * 16 + r + 8 * lhalf;
    float* crow = ctx + ((size_t)(b * S_ + qr) * H_) + h * DH_;
    crow[lmod]      = c0[r];
    crow[16 + lmod] = c1[r];
  }
}

// ---------------- fused layernorm kernels (wave per row of 256) ----------------
// MODE 0: x = a + b (attn residual)   MODE 1: x = a*sigmoid(b) + res (GRN gate)

template <int MODE>
__global__ __launch_bounds__(256) void ln_kernel(
    const float* __restrict__ A, const float* __restrict__ Bp,
    const float* __restrict__ Rp, const float* __restrict__ g,
    const float* __restrict__ be, float* __restrict__ out, int rows) {
  int wave = threadIdx.x >> 5, lane = threadIdx.x & 31;
  int row = blockIdx.x * 8 + wave;
  if (row >= rows) return;
  const size_t off = (size_t)row * 256;
  float x[8];
#pragma unroll
  for (int j = 0; j < 8; j++) {
    int e = lane + j * 32;
    float a = A[off + e];
    if (MODE == 0) {
      x[j] = a + Bp[off + e];
    } else {
      float sg = 1.f / (1.f + __expf(-Bp[off + e]));
      x[j] = a * sg + Rp[off + e];
    }
  }
  float s = 0.f;
#pragma unroll
  for (int j = 0; j < 8; j++) s += x[j];
  float m = wred(s) * (1.f / 256.f);
  float v = 0.f;
#pragma unroll
  for (int j = 0; j < 8; j++) { float d = x[j] - m; v += d * d; }
  float inv = rsqrtf(wred(v) * (1.f / 256.f) + 1e-5f);
#pragma unroll
  for (int j = 0; j < 8; j++) {
    int e = lane + j * 32;
    out[off + e] = (x[j] - m) * inv * g[e] + be[e];
  }
}

// ---------------- small elementwise kernels ----------------

__global__ __launch_bounds__(256) void pe_kernel(float* __restrict__ pe) {
  int e = blockIdx.x * 256 + threadIdx.x;  // < 512*256
  int s = e >> 8, n = e & 255;
  int j = n >> 1;
  float freq = __expf((float)(2 * j) * (-9.210340371976184f / 256.f));
  float ang = (float)s * freq;
  pe[e] = (n & 1) ? __cosf(ang) : __sinf(ang);
}

__global__ __launch_bounds__(256) void timeadd_kernel(
    float* __restrict__ Hb, const float* __restrict__ tf,
    const float* __restrict__ tpw, const float* __restrict__ tpb, int total) {
  int e = blockIdx.x * 256 + threadIdx.x;
  if (e >= total) return;
  int m = e >> 8, n = e & 255;
  float t0 = tf[m * 3], t1 = tf[m * 3 + 1], t2 = tf[m * 3 + 2];
  Hb[e] += t0 * tpw[n] + t1 * tpw[256 + n] + t2 * tpw[512 + n] + tpb[n];
}

__global__ __launch_bounds__(128) void head_kernel(
    const float* __restrict__ Hid, const float* __restrict__ W,
    const float* __restrict__ bo, float* __restrict__ out) {
  int t = threadIdx.x;  // 0..127
  int b = t >> 2, a = t & 3;
  const float* row = Hid + ((size_t)(b * 512 + 511) * 256);
  float s = bo[a];
  for (int n = 0; n < 256; n++) s += row[n] * W[n * 4 + a];
  out[b * 4 + a] = s;
}

// ---------------- orchestration ----------------

extern "C" void kernel_launch(void* const* d_in, const int* in_sizes, int n_in,
                              void* d_out, int out_size, void* d_ws, size_t ws_size,
                              hipStream_t stream) {
  (void)in_sizes; (void)n_in; (void)out_size; (void)ws_size;
  const float* x     = (const float*)d_in[0];
  const float* tf    = (const float*)d_in[1];
  const float* fp_w  = (const float*)d_in[2];
  const float* fp_b  = (const float*)d_in[3];
  const float* tp_w  = (const float*)d_in[4];
  const float* tp_b  = (const float*)d_in[5];
  const float* ip_w  = (const float*)d_in[6];
  const float* ip_b  = (const float*)d_in[7];
  const float* wq    = (const float*)d_in[8];
  const float* wk    = (const float*)d_in[9];
  const float* wv    = (const float*)d_in[10];
  const float* wo    = (const float*)d_in[11];
  const float* bq    = (const float*)d_in[12];
  const float* bk    = (const float*)d_in[13];
  const float* bv    = (const float*)d_in[14];
  const float* bo    = (const float*)d_in[15];
  const float* ln_g  = (const float*)d_in[16];
  const float* ln_b  = (const float*)d_in[17];
  const float* g_fc1w = (const float*)d_in[18];
  const float* g_fc2w = (const float*)d_in[19];
  const float* g_gw   = (const float*)d_in[20];
  const float* g_ggw  = (const float*)d_in[21];
  const float* g_fc1b = (const float*)d_in[22];
  const float* g_fc2b = (const float*)d_in[23];
  const float* g_gb   = (const float*)d_in[24];
  const float* g_ggb  = (const float*)d_in[25];
  const float* g_lg   = (const float*)d_in[26];
  const float* g_lb   = (const float*)d_in[27];
  const float* f_fc1w = (const float*)d_in[28];
  const float* f_fc2w = (const float*)d_in[29];
  const float* f_gw   = (const float*)d_in[30];
  const float* f_ggw  = (const float*)d_in[31];
  const float* f_fc1b = (const float*)d_in[32];
  const float* f_fc2b = (const float*)d_in[33];
  const float* f_gb   = (const float*)d_in[34];
  const float* f_ggb  = (const float*)d_in[35];
  const float* f_lg   = (const float*)d_in[36];
  const float* f_lb   = (const float*)d_in[37];
  const float* out_w  = (const float*)d_in[38];
  const float* out_b  = (const float*)d_in[39];

  float* outF   = (float*)d_out;
  float* logits = outF;        // 32*4
  float* aw_out = outF + 128;  // 32*8*512*512

  const int Hc = 256, Lc = 2, FINc = 199, FPKP = 224;
  const size_t TOK = (size_t)32 * 512;  // 16384
  const size_t MAT = TOK * Hc;          // 4194304 floats

  float* ws = (float*)d_ws;
  float* h  = ws;
  float* b1 = ws + MAT;
  float* b2 = ws + 2 * MAT;
  float* b3 = ws + 3 * MAT;
  float* b4 = ws + 4 * MAT;
  float* pe = ws + 5 * MAT;  // 512*256

  // bf16 transposed weight slots (16B-aligned: offset is multiple of 4 floats)
  unsigned short* wtp = (unsigned short*)(ws + 5 * MAT + 512 * 256);
  const size_t WSLOT = 256 * 256;
  unsigned short* t_fp = wtp;                 wtp += (size_t)256 * FPKP;
  unsigned short* t_ip = wtp;                 wtp += WSLOT;
  unsigned short* t_wq[2], *t_wk[2], *t_wv[2], *t_wo[2];
  unsigned short* t_g1[2], *t_g2[2], *t_gg[2], *t_gx[2];
  for (int i = 0; i < 2; i++) {
    t_wq[i] = wtp; wtp += WSLOT;  t_wk[i] = wtp; wtp += WSLOT;
    t_wv[i] = wtp; wtp += WSLOT;  t_wo[i] = wtp; wtp += WSLOT;
    t_g1[i] = wtp; wtp += WSLOT;  t_g2[i] = wtp; wtp += WSLOT;
    t_gg[i] = wtp; wtp += WSLOT;  t_gx[i] = wtp; wtp += WSLOT;
  }
  unsigned short* t_f1 = wtp; wtp += WSLOT;
  unsigned short* t_f2 = wtp; wtp += WSLOT;
  unsigned short* t_fg = wtp; wtp += WSLOT;
  unsigned short* t_fx = wtp; wtp += WSLOT;

  // convert all weights once per call (deterministic, ~2.9 MB)
  wconv_kernel<<<(256 * FPKP) / 256, 256, 0, stream>>>(fp_w, t_fp, FINc, Hc, FPKP);
  wconv_kernel<<<256, 256, 0, stream>>>(ip_w, t_ip, Hc, Hc, Hc);
  for (int i = 0; i < 2; i++) {
    size_t o = (size_t)i * Hc * Hc;
    wconv_kernel<<<256, 256, 0, stream>>>(wq + o, t_wq[i], Hc, Hc, Hc);
    wconv_kernel<<<256, 256, 0, stream>>>(wk + o, t_wk[i], Hc, Hc, Hc);
    wconv_kernel<<<256, 256, 0, stream>>>(wv + o, t_wv[i], Hc, Hc, Hc);
    wconv_kernel<<<256, 256, 0, stream>>>(wo + o, t_wo[i], Hc, Hc, Hc);
    wconv_kernel<<<256, 256, 0, stream>>>(g_fc1w + o, t_g1[i], Hc, Hc, Hc);
    wconv_kernel<<<256, 256, 0, stream>>>(g_fc2w + o, t_g2[i], Hc, Hc, Hc);
    wconv_kernel<<<256, 256, 0, stream>>>(g_gw + o, t_gg[i], Hc, Hc, Hc);
    wconv_kernel<<<256, 256, 0, stream>>>(g_ggw + o, t_gx[i], Hc, Hc, Hc);
  }
  wconv_kernel<<<256, 256, 0, stream>>>(f_fc1w, t_f1, Hc, Hc, Hc);
  wconv_kernel<<<256, 256, 0, stream>>>(f_fc2w, t_f2, Hc, Hc, Hc);
  wconv_kernel<<<256, 256, 0, stream>>>(f_gw, t_fg, Hc, Hc, Hc);
  wconv_kernel<<<256, 256, 0, stream>>>(f_ggw, t_fx, Hc, Hc, Hc);

  dim3 gG((unsigned)(TOK / 64), 2);  // 64x128 tiles over 16384x256
  dim3 bG(256);
  int lnGrid = (int)(TOK / 8);

  pe_kernel<<<512, 256, 0, stream>>>(pe);
  // input projections
  gemm_kernel<0, 0><<<gG, bG, 0, stream>>>(x, t_fp, fp_b, b1, Hc, FINc, FPKP, nullptr, 512);
  timeadd_kernel<<<(int)(MAT / 256), 256, 0, stream>>>(b1, tf, tp_w, tp_b, (int)MAT);
  gemm_kernel<0, 1><<<gG, bG, 0, stream>>>(b1, t_ip, ip_b, h, Hc, Hc, Hc, pe, 512);

  for (int i = 0; i < Lc; i++) {
    gemm_kernel<0, 0><<<gG, bG, 0, stream>>>(h, t_wq[i], bq + i * Hc, b1, Hc, Hc, Hc, nullptr, 512);
    gemm_kernel<0, 0><<<gG, bG, 0, stream>>>(h, t_wk[i], bk + i * Hc, b2, Hc, Hc, Hc, nullptr, 512);
    gemm_kernel<0, 0><<<gG, bG, 0, stream>>>(h, t_wv[i], bv + i * Hc, b3, Hc, Hc, Hc, nullptr, 512);
    attn_kernel<<<32 * 8 * 32, 32, 0, stream>>>(b1, b2, b3, b4, (i == Lc - 1) ? aw_out : nullptr);
    gemm_kernel<0, 0><<<gG, bG, 0, stream>>>(b4, t_wo[i], bo + i * Hc, b1, Hc, Hc, Hc, nullptr, 512);
    ln_kernel<0><<<lnGrid, 256, 0, stream>>>(h, b1, nullptr, ln_g + i * Hc, ln_b + i * Hc, h, (int)TOK);
    // GRN
    gemm_kernel<1, 0><<<gG, bG, 0, stream>>>(h, t_g1[i], g_fc1b + i * Hc, b1, Hc, Hc, Hc, nullptr, 512);
    gemm_kernel<1, 0><<<gG, bG, 0, stream>>>(b1, t_g2[i], g_fc2b + i * Hc, b2, Hc, Hc, Hc, nullptr, 512);
    gemm_kernel<0, 0><<<gG, bG, 0, stream>>>(b2, t_gg[i], g_gb + i * Hc, b3, Hc, Hc, Hc, nullptr, 512);
    gemm_kernel<0, 0><<<gG, bG, 0, stream>>>(b2, t_gx[i], g_ggb + i * Hc, b4, Hc, Hc, Hc, nullptr, 512);
    ln_kernel<1><<<lnGrid, 256, 0, stream>>>(b3, b4, h, g_lg + i * Hc, g_lb + i * Hc, h, (int)TOK);
  }

  // final GRN
  gemm_kernel<1, 0><<<gG, bG, 0, stream>>>(h, t_f1, f_fc1b, b1, Hc, Hc, Hc, nullptr, 512);
  gemm_kernel<1, 0><<<gG, bG, 0, stream>>>(b1, t_f2, f_fc2b, b2, Hc, Hc, Hc, nullptr, 512);
  gemm_kernel<0, 0><<<gG, bG, 0, stream>>>(b2, t_fg, f_gb, b3, Hc, Hc, Hc, nullptr, 512);
  gemm_kernel<0, 0><<<gG, bG, 0, stream>>>(b2, t_fx, f_ggb, b4, Hc, Hc, Hc, nullptr, 512);
  ln_kernel<1><<<lnGrid, 256, 0, stream>>>(b3, b4, h, f_lg, f_lb, h, (int)TOK);

  head_kernel<<<1, 128, 0, stream>>>(h, out_w, out_b, logits);
}
